// QuantLinear_54150947668073
// MI455X (gfx1250) — matrix operand order repaired and checked by reading, not compiled, not run on hardware
//
#include <hip/hip_runtime.h>

// ---------------------------------------------------------------------------
// QuantLinear: out[M,N] = X[M,K] * (W[N,K]^T * w_scale) + bias
// M = B*S = 8192, K = 4096, N = 16384 (all divisible by tile sizes)
// bf16 WMMA (V_WMMA_F32_16X16X32_BF16), fp32 accumulate.
// int8-valued weights are exact in bf16 (truncate after v_cvt_f32_i32);
// X fp32->bf16 via RNE bit-adjust; packing via v_perm_b32.
// ---------------------------------------------------------------------------

typedef __bf16 bf16_t;
typedef __attribute__((ext_vector_type(16))) __bf16 v16bf;
typedef __attribute__((ext_vector_type(8)))  __bf16 v8bf;
typedef __attribute__((ext_vector_type(8)))  float  v8f;
typedef __attribute__((ext_vector_type(4)))  float  v4f;
typedef __attribute__((ext_vector_type(4)))  int    v4i;

#define BM 128
#define BN 128
#define BK 64
#define LDA 72   /* BK + 8 bf16 pad: 144B row stride, bank-conflict-free b128 */
#define LDB 72

union FragBF {
  v16bf v;
  struct { v8bf lo, hi; } h;
};

// pack 2 fp32 -> 2 bf16 (RNE) in one dword: 2x(v_bfe+v_add3) + 1x v_perm_b32
__device__ __forceinline__ unsigned pk2_f32(float f0, float f1) {
  unsigned u0 = __builtin_bit_cast(unsigned, f0);
  unsigned u1 = __builtin_bit_cast(unsigned, f1);
  u0 += 0x7FFFu + ((u0 >> 16) & 1u);           // round-to-nearest-even
  u1 += 0x7FFFu + ((u1 >> 16) & 1u);
  // result bytes [1:0] = hi16(f0), [3:2] = hi16(f1)
  return __builtin_amdgcn_perm(u1, u0, 0x07060302u);
}

// pack 2 small ints -> 2 bf16 exactly: 2x v_cvt_f32_i32 + 1x v_perm_b32
// (|i| <= 127 fits the 8-bit bf16 significand, truncation is exact)
__device__ __forceinline__ unsigned pk2_i32(int i0, int i1) {
  unsigned u0 = __builtin_bit_cast(unsigned, (float)i0);
  unsigned u1 = __builtin_bit_cast(unsigned, (float)i1);
  return __builtin_amdgcn_perm(u1, u0, 0x07060302u);
}

__global__ __launch_bounds__(256) void qlinear_bf16_wmma(
    const float* __restrict__ X,      // [M,K] fp32
    const int*   __restrict__ W,      // [N,K] int8 values in int32
    const float* __restrict__ scale_p,// scalar
    const float* __restrict__ bias,   // [N]
    float*       __restrict__ out,    // [M,N]
    int M, int N, int K)
{
  __shared__ bf16_t sA[2][BM * LDA];
  __shared__ bf16_t sB[2][BN * LDB];

  // ---- block swizzle: sweep all N-blocks within groups of 16 M-blocks so a
  // ~32MB X supertile stays resident in the 192MB L2 while weights stream.
  int bm, bn;
  {
    const int mblocks = (int)gridDim.y;
    const int nblocks = (int)gridDim.x;
    int bid = (int)blockIdx.y * nblocks + (int)blockIdx.x;
    if ((mblocks & 15) == 0) {
      const int per = 16 * nblocks;
      const int g   = bid / per;
      const int rem = bid % per;
      bn = rem >> 4;             // rem / 16
      bm = g * 16 + (rem & 15);
    } else {
      bm = blockIdx.y; bn = blockIdx.x;
    }
  }
  const int m0 = bm * BM;
  const int n0 = bn * BN;

  const int tid  = threadIdx.x;
  const int wave = tid >> 5;        // 0..7
  const int lane = tid & 31;
  const int wm   = wave >> 2;       // 0..1 : M offset wm*64
  const int wn   = wave & 3;        // 0..3 : N offset wn*32
  const int r    = lane & 15;
  const int h    = lane >> 4;       // half-wave selector

  v8f acc[4][2] = {};               // 4 M-tiles x 2 N-tiles of 16x16 f32

  // ---- prologue: stage k-tile 0 into buffer 0 (convert fp32/int32 -> bf16)
  {
    #pragma unroll
    for (int t = 0; t < 8; ++t) {
      const int v   = t * 256 + tid;      // 2048 vec4 slots: 128 rows x 16
      const int row = v >> 4;
      const int c4  = v & 15;
      v4f a = *(const v4f*)(X + (size_t)(m0 + row) * K + c4 * 4);
      v4i b = *(const v4i*)(W + (size_t)(n0 + row) * K + c4 * 4);
      uint2 ap, bp;
      ap.x = pk2_f32(a[0], a[1]); ap.y = pk2_f32(a[2], a[3]);
      bp.x = pk2_i32(b[0], b[1]); bp.y = pk2_i32(b[2], b[3]);
      *(uint2*)&sA[0][row * LDA + c4 * 4] = ap;
      *(uint2*)&sB[0][row * LDB + c4 * 4] = bp;
    }
  }

  int buf = 0;
  for (int kt = 0; kt < K; kt += BK) {
    const bool more = (kt + BK) < K;
    v4f pa[8]; v4i pb[8];

    if (more) {
      // issue global loads for the next k-tile (overlap with compute)
      #pragma unroll
      for (int t = 0; t < 8; ++t) {
        const int v   = t * 256 + tid;
        const int row = v >> 4;
        const int c4  = v & 15;
        pa[t] = *(const v4f*)(X + (size_t)(m0 + row) * K + (kt + BK) + c4 * 4);
        pb[t] = *(const v4i*)(W + (size_t)(n0 + row) * K + (kt + BK) + c4 * 4);
      }
      if (kt + 2 * BK < K) {
        // pull tile k+2 into L2 while the matrix cores run
        const int prow = tid >> 1;
        const int pc   = (tid & 1) * 32;
        __builtin_prefetch(X + (size_t)(m0 + prow) * K + kt + 2 * BK + pc, 0, 1);
        __builtin_prefetch(W + (size_t)(n0 + prow) * K + kt + 2 * BK + pc, 0, 1);
      }
    }

    __syncthreads();   // staged buf visible; prior reads of buf^1 retired

    // ---- compute: 2 K-steps of 32, 8 WMMAs each
    #pragma unroll
    for (int ks = 0; ks < 2; ++ks) {
      FragBF af[4], bfr[2];
      #pragma unroll
      for (int i = 0; i < 4; ++i) {
        // A 16x32 bf16: lane<16 -> K {0..7,16..23}; lane>=16 -> {8..15,24..31}
        const bf16_t* p = &sA[buf][(wm * 64 + i * 16 + r) * LDA + ks * 32 + h * 8];
        af[i].h.lo = *(const v8bf*)p;
        af[i].h.hi = *(const v8bf*)(p + 16);
      }
      #pragma unroll
      for (int j = 0; j < 2; ++j) {
        // B 32x16 bf16: lane<16 -> K {0..7,16..23}; lane>=16 -> {8..15,24..31}
        const bf16_t* p = &sB[buf][(wn * 32 + j * 16 + r) * LDB + ks * 32 + h * 8];
        bfr[j].h.lo = *(const v8bf*)p;
        bfr[j].h.hi = *(const v8bf*)(p + 16);
      }
      #pragma unroll
      for (int i = 0; i < 4; ++i)
        #pragma unroll
        for (int j = 0; j < 2; ++j)
          acc[i][j] = __builtin_amdgcn_wmma_f32_16x16x32_bf16(
              false, af[i].v, false, bfr[j].v,
              (short)0, acc[i][j], false, false);
    }

    if (more) {
      // convert + store prefetched tile into the other buffer
      #pragma unroll
      for (int t = 0; t < 8; ++t) {
        const int v   = t * 256 + tid;
        const int row = v >> 4;
        const int c4  = v & 15;
        uint2 ap, bp;
        ap.x = pk2_f32(pa[t][0], pa[t][1]); ap.y = pk2_f32(pa[t][2], pa[t][3]);
        bp.x = pk2_i32(pb[t][0], pb[t][1]); bp.y = pk2_i32(pb[t][2], pb[t][3]);
        *(uint2*)&sA[buf ^ 1][row * LDA + c4 * 4] = ap;
        *(uint2*)&sB[buf ^ 1][row * LDB + c4 * 4] = bp;
      }
    }
    buf ^= 1;
  }

  // ---- epilogue: scale + bias, fp32 stores
  const float sc = scale_p[0];
  #pragma unroll
  for (int j = 0; j < 2; ++j) {
    const int   n  = n0 + wn * 32 + j * 16 + r;
    const float bv = bias[n];
    #pragma unroll
    for (int i = 0; i < 4; ++i) {
      const int mb = m0 + wm * 64 + i * 16 + h * 8;
      #pragma unroll
      for (int e = 0; e < 8; ++e)
        out[(size_t)(mb + e) * N + n] = acc[i][j][e] * sc + bv;
    }
  }
}

extern "C" void kernel_launch(void* const* d_in, const int* in_sizes, int n_in,
                              void* d_out, int out_size, void* d_ws, size_t ws_size,
                              hipStream_t stream) {
  (void)n_in; (void)out_size; (void)d_ws; (void)ws_size;
  const float* X     = (const float*)d_in[0];   // [B,S,D_IN] fp32
  const int*   W     = (const int*)  d_in[1];   // [D_OUT,D_IN] int32 (int8 values)
  const float* scale = (const float*)d_in[2];   // scalar
  const float* bias  = (const float*)d_in[3];   // [D_OUT]

  const int N = in_sizes[3];                    // D_OUT
  const int K = in_sizes[1] / N;                // D_IN
  const int M = in_sizes[0] / K;                // B*S

  dim3 grid(N / BN, M / BM);                    // (128, 64)
  qlinear_bf16_wmma<<<grid, dim3(256, 1, 1), 0, stream>>>(
      X, W, scale, bias, (float*)d_out, M, N, K);
}
